// LatentMatrixMemory_24215025615154
// MI455X (gfx1250) — matrix operand order repaired and checked by reading, not compiled, not run on hardware
//
#include <hip/hip_runtime.h>
#include <stdint.h>

#define BB       64
#define TT       2048
#define DKK      128
#define DVV      128
#define DVWW     64     // value columns per workgroup (DV split in 2)
#define CC       32     // chunk length
#define NTHREADS 256    // 8 waves (wave32)

// ---- padded LDS strides (element units) ----
#define ST_MT 132   // f32, rows of 128 (+4)
#define ST_MB 136   // bf16, rows of 128 (+8)
#define ST_Q  136   // bf16 Qs / Kh rows of 128 (+8)
#define ST_K2 40    // bf16 Ks2T rows of 32 (+8)
#define ST_V  40    // bf16 VbT rows of 32 (+8)
#define ST_AS 40    // bf16 As rows of 32 (+8)

// ---- LDS layout (byte offsets, all 16B aligned) ----
#define OFF_MT 0
#define SZ_MT  (64 * ST_MT * 4)          // 33792
#define OFF_MB (OFF_MT + SZ_MT)          // 33792
#define SZ_MB  (64 * ST_MB * 2)          // 17408
#define OFF_QS (OFF_MB + SZ_MB)          // 51200
#define SZ_QS  (32 * ST_Q * 2)           // 8704
#define OFF_KH (OFF_QS + SZ_QS)          // 59904
#define SZ_KH  (32 * ST_Q * 2)           // 8704
#define OFF_K2 (OFF_KH + SZ_KH)          // 68608
#define SZ_K2  (128 * ST_K2 * 2)         // 10240
#define OFF_VB (OFF_K2 + SZ_K2)          // 78848
#define SZ_VB  (64 * ST_V * 2)           // 5120
#define OFF_AS (OFF_VB + SZ_VB)          // 83968
#define SZ_AS  (32 * ST_AS * 2)          // 2560
#define OFF_SC (OFF_AS + SZ_AS)          // 86528
#define SZ_SC  ((6 * 32 + 4) * 4)        // 784
#define SMEM_BYTES (OFF_SC + SZ_SC)      // 87312

typedef __attribute__((ext_vector_type(16))) __bf16       v16bf;
typedef __attribute__((ext_vector_type(8)))  float        v8f;
typedef __attribute__((ext_vector_type(4)))  float        vf4;
typedef __attribute__((ext_vector_type(4)))  unsigned int vu4;

union BFV { v16bf bf; vu4 u[2]; };

__device__ __forceinline__ unsigned short f2bf(float f) {
  unsigned int u = __float_as_uint(f);
  u += 0x7FFFu + ((u >> 16) & 1u);   // round-to-nearest-even
  return (unsigned short)(u >> 16);
}

__device__ __forceinline__ v8f wmma_bf16(v16bf a, v16bf b, v8f c) {
  // (neg_a, A, neg_b, B, c_mod, C, reuse_a, reuse_b)
  return __builtin_amdgcn_wmma_f32_16x16x32_bf16(false, a, false, b, (short)0, c,
                                                 false, false);
}

// A-matrix tile (16x32 bf16) from row-major [m][k] LDS array.
// ISA layout: lane L (<16): row M=L, K = kb+{0..7} and kb+{16..23}, kb=(L/16)*8.
__device__ __forceinline__ v16bf load_a(const unsigned short* p, int stride,
                                        int row0, int k0, int lane) {
  int r  = row0 + (lane & 15);
  int kb = k0 + ((lane >> 4) << 3);
  BFV t;
  t.u[0] = *(const vu4*)(p + r * stride + kb);
  t.u[1] = *(const vu4*)(p + r * stride + kb + 16);
  return t.bf;
}

// B-matrix tile (32x16 bf16): B[k][n] = p[n][k], p stored n-major.
// ISA layout: lane L holds N=L%16, K = kb+{0..7} and kb+{16..23}, kb=(L/16)*8.
__device__ __forceinline__ v16bf load_b(const unsigned short* p, int stride,
                                        int n0, int k0, int lane) {
  int n  = n0 + (lane & 15);
  int kb = k0 + ((lane >> 4) << 3);
  BFV t;
  t.u[0] = *(const vu4*)(p + n * stride + kb);
  t.u[1] = *(const vu4*)(p + n * stride + kb + 16);
  return t.bf;
}

__global__ __launch_bounds__(NTHREADS, 1)
void latent_matrix_memory_kernel(const float* __restrict__ q,
                                 const float* __restrict__ k,
                                 const float* __restrict__ v,
                                 const float* __restrict__ alpha,
                                 const float* __restrict__ rho,
                                 const float* __restrict__ lam,
                                 const float* __restrict__ init,
                                 float* __restrict__ out) {
  extern __shared__ char smem[];
  float*          MT   = (float*)(smem + OFF_MT);            // [64][ST_MT]  state^T (f32)
  unsigned short* MbT  = (unsigned short*)(smem + OFF_MB);   // [64][ST_MB]  state^T (bf16)
  unsigned short* Qs   = (unsigned short*)(smem + OFF_QS);   // [32][ST_Q]   P_t * q_t
  unsigned short* Kh   = (unsigned short*)(smem + OFF_KH);   // [32][ST_Q]   (w/P) * k
  unsigned short* K2T  = (unsigned short*)(smem + OFF_K2);   // [128][ST_K2] ((Plast/P)w k)^T
  unsigned short* VbT  = (unsigned short*)(smem + OFF_VB);   // [64][ST_V]   v^T
  unsigned short* As   = (unsigned short*)(smem + OFF_AS);   // [32][ST_AS]  masked scores
  float* sQ   = (float*)(smem + OFF_SC);
  float* sKh  = sQ + 32;
  float* sK2  = sQ + 64;
  float* araw = sQ + 96;
  float* rraw = sQ + 128;
  float* lraw = sQ + 160;
  float* plsh = sQ + 192;

  const int tid   = threadIdx.x;
  const int lane  = tid & 31;
  const int wave  = tid >> 5;
  const int b     = blockIdx.x >> 1;
  const int slice = blockIdx.x & 1;

  const float* qb = q    + (size_t)b * TT * DKK;
  const float* kb = k    + (size_t)b * TT * DKK;
  const float* vb = v    + (size_t)b * TT * DVV + slice * DVWW;
  const float* ab = alpha + (size_t)b * TT;
  const float* rb = rho   + (size_t)b * TT;
  const float* lb = lam   + (size_t)b * TT;
  const float* ib = init + (size_t)b * DKK * DVV + slice * DVWW;
  float*       ob = out  + (size_t)b * TT * DVV + slice * DVWW;

  // ---- init: MT[n][m] = init_state[b][m][slice*64+n]; zero score buffer ----
  for (int i = tid; i < DKK * DVWW; i += NTHREADS) {
    int m = i >> 6;        // d
    int n = i & 63;        // vcol
    MT[n * ST_MT + m] = ib[(size_t)m * DVV + n];
  }
  for (int i = tid; i < CC * ST_AS; i += NTHREADS) As[i] = 0;
  __syncthreads();

  for (int ch = 0; ch < TT / CC; ++ch) {
    const int t0 = ch * CC;

    // ---- phase a: per-step scalars + serial decay scan (32 steps) ----
    if (tid < CC) {
      araw[tid] = ab[t0 + tid];
      rraw[tid] = rb[t0 + tid];
      lraw[tid] = lb[t0 + tid];
    }
    __syncthreads();
    if (tid == 0) {
      float P = 1.0f;
      for (int t = 0; t < CC; ++t) {
        P *= lraw[t];
        float w = rraw[t] * araw[t];
        sQ[t]  = P;        // scale for q_t
        sKh[t] = w / P;    // scale for intra-chunk keys
      }
      plsh[0] = P;
      for (int t = 0; t < CC; ++t) sK2[t] = P * sKh[t];  // scale for state keys
    }
    __syncthreads();
    const float pls = plsh[0];

    // ---- phase b: load chunk, scale, convert to bf16 staging ----
    for (int i = tid; i < (CC * DKK) / 4; i += NTHREADS) {   // Qs
      int t = i >> 5, d4 = (i & 31) << 2;
      vf4 x = *(const vf4*)(qb + (size_t)(t0 + t) * DKK + d4);
      float s = sQ[t];
      unsigned short* dst = Qs + t * ST_Q + d4;
      dst[0] = f2bf(x.x * s); dst[1] = f2bf(x.y * s);
      dst[2] = f2bf(x.z * s); dst[3] = f2bf(x.w * s);
    }
    for (int i = tid; i < (CC * DKK) / 4; i += NTHREADS) {   // Kh + K2T
      int t = i >> 5, d4 = (i & 31) << 2;
      vf4 x = *(const vf4*)(kb + (size_t)(t0 + t) * DKK + d4);
      float s1 = sKh[t], s2 = sK2[t];
      unsigned short* dst = Kh + t * ST_Q + d4;
      dst[0] = f2bf(x.x * s1); dst[1] = f2bf(x.y * s1);
      dst[2] = f2bf(x.z * s1); dst[3] = f2bf(x.w * s1);
      K2T[(d4 + 0) * ST_K2 + t] = f2bf(x.x * s2);
      K2T[(d4 + 1) * ST_K2 + t] = f2bf(x.y * s2);
      K2T[(d4 + 2) * ST_K2 + t] = f2bf(x.z * s2);
      K2T[(d4 + 3) * ST_K2 + t] = f2bf(x.w * s2);
    }
    for (int i = tid; i < (CC * DVWW) / 4; i += NTHREADS) {  // VbT
      int t = i >> 4, j4 = (i & 15) << 2;
      vf4 x = *(const vf4*)(vb + (size_t)(t0 + t) * DVV + j4);
      VbT[(j4 + 0) * ST_V + t] = f2bf(x.x);
      VbT[(j4 + 1) * ST_V + t] = f2bf(x.y);
      VbT[(j4 + 2) * ST_V + t] = f2bf(x.z);
      VbT[(j4 + 3) * ST_V + t] = f2bf(x.w);
    }
    for (int i = tid; i < (DVWW * DKK) / 4; i += NTHREADS) { // MT -> MbT (bf16)
      int n = i >> 5, m4 = (i & 31) << 2;
      vf4 x = *(const vf4*)(MT + n * ST_MT + m4);
      unsigned short* dst = MbT + n * ST_MB + m4;
      dst[0] = f2bf(x.x); dst[1] = f2bf(x.y);
      dst[2] = f2bf(x.z); dst[3] = f2bf(x.w);
    }
    // prefetch next chunk (global_prefetch_b8)
    if (ch + 1 < TT / CC) {
      __builtin_prefetch(qb + (size_t)(t0 + CC) * DKK + tid * 16, 0, 1);
      __builtin_prefetch(kb + (size_t)(t0 + CC) * DKK + tid * 16, 0, 1);
      if (tid < 64)
        __builtin_prefetch(vb + (size_t)(t0 + CC + (tid >> 1)) * DVV + (tid & 1) * 32, 0, 1);
    }
    __syncthreads();

    // ---- phase c: causal scores A = tril(Qs @ Kh^T), 3 non-trivial 16x16 tiles ----
    if (wave < 3) {
      const int ti = (wave == 0) ? 0 : 1;
      const int tj = (wave == 2) ? 1 : 0;
      v8f c = {0.f, 0.f, 0.f, 0.f, 0.f, 0.f, 0.f, 0.f};
      #pragma unroll
      for (int kt = 0; kt < 4; ++kt) {
        v16bf a  = load_a(Qs, ST_Q, ti * 16, kt * 32, lane);
        v16bf bm = load_b(Kh, ST_Q, tj * 16, kt * 32, lane);
        c = wmma_bf16(a, bm, c);
      }
      const int n  = lane & 15;
      const int m0 = (lane >> 4) << 3;
      const int sl = tj * 16 + n;
      #pragma unroll
      for (int r = 0; r < 8; ++r) {
        int tl = ti * 16 + m0 + r;
        float val = (sl <= tl) ? c[r] : 0.f;
        As[tl * ST_AS + sl] = f2bf(val);
      }
    }
    __syncthreads();

    // ---- phase d: readout (inter + intra), one 16x16 out tile per wave ----
    {
      const int ti = wave >> 2, vt = wave & 3;
      v8f c = {0.f, 0.f, 0.f, 0.f, 0.f, 0.f, 0.f, 0.f};
      #pragma unroll
      for (int kt = 0; kt < 4; ++kt) {                       // (P q) @ M0
        v16bf a  = load_a(Qs, ST_Q, ti * 16, kt * 32, lane);
        v16bf bm = load_b(MbT, ST_MB, vt * 16, kt * 32, lane);
        c = wmma_bf16(a, bm, c);
      }
      {                                                      // tril(scores) @ V
        v16bf a  = load_a(As, ST_AS, ti * 16, 0, lane);
        v16bf bm = load_b(VbT, ST_V, vt * 16, 0, lane);
        c = wmma_bf16(a, bm, c);
      }
      const int n  = lane & 15;
      const int m0 = (lane >> 4) << 3;
      const int vg = vt * 16 + n;
      #pragma unroll
      for (int r = 0; r < 8; ++r) {
        int tl = ti * 16 + m0 + r;
        ob[(size_t)(t0 + tl) * DVV + vg] = c[r];
      }
    }

    // ---- phase e: state update M = Plast*M + K2^T @ V (f32 accum in WMMA C) ----
    {
      const int dt = wave;                                   // 16 d-rows per wave
      const int n  = lane & 15;
      const int m0 = (lane >> 4) << 3;
      #pragma unroll
      for (int vt = 0; vt < 4; ++vt) {
        float* cp = MT + (vt * 16 + n) * ST_MT + dt * 16 + m0;
        vf4 c0 = *(vf4*)cp;
        vf4 c1 = *(vf4*)(cp + 4);
        v8f c = {c0.x * pls, c0.y * pls, c0.z * pls, c0.w * pls,
                 c1.x * pls, c1.y * pls, c1.z * pls, c1.w * pls};
        v16bf a  = load_a(K2T, ST_K2, dt * 16, 0, lane);
        v16bf bm = load_b(VbT, ST_V, vt * 16, 0, lane);
        c = wmma_bf16(a, bm, c);
        vf4 o0 = {c[0], c[1], c[2], c[3]};
        vf4 o1 = {c[4], c[5], c[6], c[7]};
        *(vf4*)cp       = o0;
        *(vf4*)(cp + 4) = o1;
      }
    }
    __syncthreads();   // protect staging + MT for next chunk
  }
}

extern "C" void kernel_launch(void* const* d_in, const int* in_sizes, int n_in,
                              void* d_out, int out_size, void* d_ws, size_t ws_size,
                              hipStream_t stream) {
  (void)in_sizes; (void)n_in; (void)out_size; (void)d_ws; (void)ws_size;
  const float* q     = (const float*)d_in[0];
  const float* k     = (const float*)d_in[1];
  const float* v     = (const float*)d_in[2];
  const float* alpha = (const float*)d_in[3];
  const float* rho   = (const float*)d_in[4];
  const float* lam   = (const float*)d_in[5];
  const float* init  = (const float*)d_in[6];
  float* out = (float*)d_out;

  hipFuncSetAttribute((const void*)latent_matrix_memory_kernel,
                      hipFuncAttributeMaxDynamicSharedMemorySize,
                      (int)SMEM_BYTES);

  dim3 grid(BB * 2);      // 64 batches x 2 value-column slices
  dim3 block(NTHREADS);   // 8 wave32 waves
  latent_matrix_memory_kernel<<<grid, block, SMEM_BYTES, stream>>>(
      q, k, v, alpha, rho, lam, init, out);
}
